// FeaStNetwork_50371376447772
// MI455X (gfx1250) — matrix operand order repaired and checked by reading, not compiled, not run on hardware
//
#include <hip/hip_runtime.h>
#include <hip/hip_bf16.h>
#include <math.h>

// ---------------------------------------------------------------------------
// FeaStNet forward for MI455X (gfx1250, wave32, WMMA).
//
// Pipeline (all intermediates f32 in d_ws):
//   h0  = elu(x @ fc0_w + b)                       [N,32]
//   XW  = h  @ conv_w   (WMMA f16->f32 GEMM)       [N,H*out]
//   XU  = h  @ conv_u                              [N,4]
//   edge pass: q = softmax(XU[s]-XU[d]+c); atomically scatter
//              sum_h q_h * XW[s,h,:] into agg[d]   (L2-resident gathers)
//   h'  = elu(agg/cnt + b)
//   h3  = elu(h2 @ fc1_w + b)  (WMMA, fused bias+ELU)
//   out = log_softmax(h3 @ fc2_w + b)  (wave-per-node, shfl reductions)
// ---------------------------------------------------------------------------

typedef _Float16 half8  __attribute__((ext_vector_type(8)));
typedef _Float16 half16 __attribute__((ext_vector_type(16)));
typedef float    float8 __attribute__((ext_vector_type(8)));

static constexpr int F_IN = 3, CH = 32, LIN = 256, NCLS = 25;

__device__ __forceinline__ float elu1(float v) { return v > 0.f ? v : expm1f(v); }

// ---------------- fc0: h0 = elu(x @ W[3,32] + b) ---------------------------
__global__ void fc0_elu_kernel(const float* __restrict__ x,
                               const float* __restrict__ w,
                               const float* __restrict__ b,
                               float* __restrict__ h0, int n_nodes) {
  int idx = blockIdx.x * blockDim.x + threadIdx.x;
  if (idx >= n_nodes * CH) return;
  int node = idx / CH, c = idx - node * CH;
  float acc = b[c];
#pragma unroll
  for (int k = 0; k < F_IN; ++k) acc += x[node * F_IN + k] * w[k * CH + c];
  h0[idx] = elu1(acc);
}

// ---------------- degree (incl. self loop) ---------------------------------
__global__ void degree_kernel(const long long* __restrict__ ei, int e_raw,
                              int n_nodes, float* __restrict__ cnt) {
  int e = blockIdx.x * blockDim.x + threadIdx.x;
  if (e >= e_raw + n_nodes) return;
  int d = (e < e_raw) ? (int)ei[(size_t)e_raw + e] : (e - e_raw);
  atomicAdd(&cnt[d], 1.0f);
}

// ---------------- XU = h @ u[K,4]  (per-node attention projections) --------
__global__ void node_u_kernel(const float* __restrict__ h,
                              const float* __restrict__ u,
                              float* __restrict__ xu, int n_nodes, int K) {
  int n = blockIdx.x * blockDim.x + threadIdx.x;
  if (n >= n_nodes) return;
  float a0 = 0.f, a1 = 0.f, a2 = 0.f, a3 = 0.f;
  const float* hr = h + (size_t)n * K;
  for (int k = 0; k < K; ++k) {
    float v = hr[k];
    a0 += v * u[k * 4 + 0]; a1 += v * u[k * 4 + 1];
    a2 += v * u[k * 4 + 2]; a3 += v * u[k * 4 + 3];
  }
  xu[(size_t)n * 4 + 0] = a0; xu[(size_t)n * 4 + 1] = a1;
  xu[(size_t)n * 4 + 2] = a2; xu[(size_t)n * 4 + 3] = a3;
}

// ---------------- WMMA GEMM: C[n_rows,M] = A[n_rows,K] @ B[K,M] ------------
// 256 threads = 8 waves; block tile 32 rows x 64 cols; wave tile 16x16.
// K in {32,64,128} (template -> fully unrolled v_wmma loop).
template <int K>
__global__ __launch_bounds__(256)
void gemm_wmma_kernel(const float* __restrict__ A, const float* __restrict__ B,
                      const float* __restrict__ bias, float* __restrict__ C,
                      int n_rows, int M, int act) {
  constexpr int AST = K + 8;  // padded LDS strides (f16 units): 16B-aligned rows,
  constexpr int BST = K + 8;  // bank-conflict-free 4-dword fragment reads
  __shared__ __align__(16) _Float16 As[32 * AST];
  __shared__ __align__(16) _Float16 Bt[64 * BST];  // B transposed: [col][k]

  const int tid = threadIdx.x;
  const int row0 = blockIdx.x * 32;
  const int col0 = blockIdx.y * 64;

  // Stage A tile (f32->f16), zero-fill out-of-range rows.
  for (int idx = tid; idx < 32 * K; idx += 256) {
    int r = idx / K, k = idx - r * K;
    int row = row0 + r;
    float v = (row < n_rows) ? A[(size_t)row * K + k] : 0.f;
    As[r * AST + k] = (_Float16)v;
  }
  // Stage B tile transposed (coalesced global reads along M).
  for (int idx = tid; idx < 64 * K; idx += 256) {
    int nn = idx & 63, k = idx >> 6;
    Bt[nn * BST + k] = (_Float16)B[(size_t)k * M + col0 + nn];
  }
  __syncthreads();

  const int wave = tid >> 5, lane = tid & 31;
  const int half = lane >> 4;            // ISA A/B fragment half-wave select
  const int rtile = wave >> 2, ctile = wave & 3;
  const _Float16* arow = &As[(rtile * 16 + (lane & 15)) * AST];
  const _Float16* brow = &Bt[(ctile * 16 + (lane & 15)) * BST];

  float8 acc = {};
#pragma unroll
  for (int kk = 0; kk < K; kk += 32) {
    // A 16x32 f16 fragment: elems 0-7 -> K = half*8+{0..7}; 8-15 -> 16+half*8+{0..7}
    half8 alo = *(const half8*)(arow + kk + half * 8);
    half8 ahi = *(const half8*)(arow + kk + 16 + half * 8);
    // B 32x16 fragment: lane's column, elems 0-7 -> K = half*8+{0..7}; 8-15 -> 16+half*8+{0..7}
    half8 blo = *(const half8*)(brow + kk + half * 8);
    half8 bhi = *(const half8*)(brow + kk + 16 + half * 8);
    half16 a = __builtin_shufflevector(alo, ahi, 0, 1, 2, 3, 4, 5, 6, 7,
                                       8, 9, 10, 11, 12, 13, 14, 15);
    half16 b = __builtin_shufflevector(blo, bhi, 0, 1, 2, 3, 4, 5, 6, 7,
                                       8, 9, 10, 11, 12, 13, 14, 15);
    acc = __builtin_amdgcn_wmma_f32_16x16x32_f16(false, a, false, b,
                                                 (short)0, acc, false, false);
  }

  // D layout: elem r -> row = 8*half + r, col = lane&15 (within 16x16 tile)
  const int col = col0 + ctile * 16 + (lane & 15);
  const float bv = bias ? bias[col] : 0.f;
#pragma unroll
  for (int r = 0; r < 8; ++r) {
    int row = row0 + rtile * 16 + half * 8 + r;
    if (row < n_rows) {
      float v = acc[r] + bv;
      if (act) v = elu1(v);
      C[(size_t)row * M + col] = v;
    }
  }
}

// ---------------- FeaSt edge pass: one wave32 per edge ---------------------
__global__ __launch_bounds__(256)
void feast_edge_kernel(const long long* __restrict__ ei, int e_raw, int n_nodes,
                       const float* __restrict__ xu, const float* __restrict__ hc,
                       const float* __restrict__ xw, float* __restrict__ agg,
                       int OUT) {
  const int wid = (blockIdx.x * blockDim.x + threadIdx.x) >> 5;
  const int lane = threadIdx.x & 31;
  if (wid >= e_raw + n_nodes) return;
  int s, d;
  if (wid < e_raw) { s = (int)ei[wid]; d = (int)ei[(size_t)e_raw + wid]; }
  else             { s = d = wid - e_raw; }

  const float* w0 = xw + (size_t)s * 4 * OUT;
  if (lane * 16 < 4 * OUT)                 // pull the XW row toward us early
    __builtin_prefetch(w0 + lane * 16, 0, 0);

  // q = softmax( XU[s] - XU[d] + c )  over 4 heads (broadcast loads, cheap)
  const float* xs = xu + (size_t)s * 4;
  const float* xd = xu + (size_t)d * 4;
  float l0 = xs[0] - xd[0] + hc[0], l1 = xs[1] - xd[1] + hc[1];
  float l2 = xs[2] - xd[2] + hc[2], l3 = xs[3] - xd[3] + hc[3];
  float mx = fmaxf(fmaxf(l0, l1), fmaxf(l2, l3));
  float e0 = expf(l0 - mx), e1 = expf(l1 - mx);
  float e2 = expf(l2 - mx), e3 = expf(l3 - mx);
  float inv = 1.f / (e0 + e1 + e2 + e3);
  float q0 = e0 * inv, q1 = e1 * inv, q2 = e2 * inv, q3 = e3 * inv;

  for (int j = lane; j < OUT; j += 32) {   // coalesced gather + f32 atomic scatter
    float m = q0 * w0[j] + q1 * w0[OUT + j] + q2 * w0[2 * OUT + j] +
              q3 * w0[3 * OUT + j];
    atomicAdd(&agg[(size_t)d * OUT + j], m);
  }
}

// ---------------- h' = elu(agg/cnt + b) ------------------------------------
__global__ void finalize_kernel(const float* __restrict__ agg,
                                const float* __restrict__ cnt,
                                const float* __restrict__ b,
                                float* __restrict__ h, int n_nodes, int OUT) {
  int idx = blockIdx.x * blockDim.x + threadIdx.x;
  if (idx >= n_nodes * OUT) return;
  int n = idx / OUT, j = idx - n * OUT;
  h[idx] = elu1(agg[idx] / cnt[n] + b[j]);
}

// ---------------- fc2 + log_softmax: one wave32 per node -------------------
__global__ __launch_bounds__(256)
void fc2_logsoftmax_kernel(const float* __restrict__ h3,
                           const float* __restrict__ w,
                           const float* __restrict__ bias,
                           float* __restrict__ out, int n_nodes) {
  __shared__ float sh[8][LIN];
  const int wave = threadIdx.x >> 5, lane = threadIdx.x & 31;
  const int n = blockIdx.x * 8 + wave;
  const bool live = n < n_nodes;
  const int nn = live ? n : 0;
#pragma unroll
  for (int i = 0; i < LIN / 32; ++i)
    sh[wave][lane + 32 * i] = h3[(size_t)nn * LIN + lane + 32 * i];
  __syncthreads();

  float dot = -INFINITY;
  if (lane < NCLS) {
    float a = bias[lane];
    for (int k = 0; k < LIN; ++k) a += sh[wave][k] * w[k * NCLS + lane];
    dot = a;
  }
  float mx = dot;
#pragma unroll
  for (int off = 16; off > 0; off >>= 1) mx = fmaxf(mx, __shfl_xor(mx, off, 32));
  float ex = (lane < NCLS) ? expf(dot - mx) : 0.f;
  float sum = ex;
#pragma unroll
  for (int off = 16; off > 0; off >>= 1) sum += __shfl_xor(sum, off, 32);
  if (live && lane < NCLS)
    out[(size_t)n * NCLS + lane] = dot - mx - logf(sum);
}

// ---------------------------------------------------------------------------
extern "C" void kernel_launch(void* const* d_in, const int* in_sizes, int n_in,
                              void* d_out, int out_size, void* d_ws, size_t ws_size,
                              hipStream_t stream) {
  const float*     x     = (const float*)d_in[0];
  const long long* ei    = (const long long*)d_in[1];
  const float* fc0_w = (const float*)d_in[2];  const float* fc0_b = (const float*)d_in[3];
  const float* c1_w  = (const float*)d_in[4];  const float* c1_u  = (const float*)d_in[5];
  const float* c1_c  = (const float*)d_in[6];  const float* c1_b  = (const float*)d_in[7];
  const float* c2_w  = (const float*)d_in[8];  const float* c2_u  = (const float*)d_in[9];
  const float* c2_c  = (const float*)d_in[10]; const float* c2_b  = (const float*)d_in[11];
  const float* fc1_w = (const float*)d_in[12]; const float* fc1_b = (const float*)d_in[13];
  const float* fc2_w = (const float*)d_in[14]; const float* fc2_b = (const float*)d_in[15];

  const int n     = in_sizes[0] / F_IN;   // 50000
  const int e_raw = in_sizes[1] / 2;      // 800000
  const int tot_e = e_raw + n;            // edges + self loops

  // workspace carve-out (f32): ~N*1125 floats total (~225 MB)
  float* ws  = (float*)d_ws;
  float* cnt = ws;                          // N
  float* h0  = cnt + n;                     // N*32
  float* xw  = h0  + (size_t)n * 32;        // N*512 (reused conv1: N*256)
  float* xu  = xw  + (size_t)n * 512;       // N*4
  float* agg = xu  + (size_t)n * 4;         // N*128 (reused conv1: N*64)
  float* h1  = agg + (size_t)n * 128;       // N*64
  float* h2  = h1  + (size_t)n * 64;        // N*128
  float* h3  = h2  + (size_t)n * 128;       // N*256
  float* outp = (float*)d_out;

  (void)hipMemsetAsync(cnt, 0, (size_t)n * sizeof(float), stream);
  degree_kernel<<<(tot_e + 255) / 256, 256, 0, stream>>>(ei, e_raw, n, cnt);
  fc0_elu_kernel<<<(n * CH + 255) / 256, 256, 0, stream>>>(x, fc0_w, fc0_b, h0, n);

  // ---- conv1: in=32, out=64, H=4 -> XW [N,256]
  {
    dim3 g((n + 31) / 32, 256 / 64);
    gemm_wmma_kernel<32><<<g, 256, 0, stream>>>(h0, c1_w, nullptr, xw, n, 256, 0);
  }
  node_u_kernel<<<(n + 255) / 256, 256, 0, stream>>>(h0, c1_u, xu, n, 32);
  (void)hipMemsetAsync(agg, 0, (size_t)n * 64 * sizeof(float), stream);
  feast_edge_kernel<<<(tot_e + 7) / 8, 256, 0, stream>>>(ei, e_raw, n, xu, c1_c, xw, agg, 64);
  finalize_kernel<<<(n * 64 + 255) / 256, 256, 0, stream>>>(agg, cnt, c1_b, h1, n, 64);

  // ---- conv2: in=64, out=128, H=4 -> XW [N,512]
  {
    dim3 g((n + 31) / 32, 512 / 64);
    gemm_wmma_kernel<64><<<g, 256, 0, stream>>>(h1, c2_w, nullptr, xw, n, 512, 0);
  }
  node_u_kernel<<<(n + 255) / 256, 256, 0, stream>>>(h1, c2_u, xu, n, 64);
  (void)hipMemsetAsync(agg, 0, (size_t)n * 128 * sizeof(float), stream);
  feast_edge_kernel<<<(tot_e + 7) / 8, 256, 0, stream>>>(ei, e_raw, n, xu, c2_c, xw, agg, 128);
  finalize_kernel<<<(n * 128 + 255) / 256, 256, 0, stream>>>(agg, cnt, c2_b, h2, n, 128);

  // ---- fc1 (fused bias + ELU)
  {
    dim3 g((n + 31) / 32, 256 / 64);
    gemm_wmma_kernel<128><<<g, 256, 0, stream>>>(h2, fc1_w, fc1_b, h3, n, 256, 1);
  }

  // ---- fc2 + log_softmax
  fc2_logsoftmax_kernel<<<(n + 7) / 8, 256, 0, stream>>>(h3, fc2_w, fc2_b, outp, n);
}
